// MatcherPair_33328946217021
// MI455X (gfx1250) — matrix operand order repaired and checked by reading, not compiled, not run on hardware
//
#include <hip/hip_runtime.h>
#include <math.h>

// ---------------------------------------------------------------------------
// CDNA5 (gfx1250) types / WMMA helpers: wave32, V_WMMA_F32_16X16X32_BF16
// ---------------------------------------------------------------------------
typedef __attribute__((ext_vector_type(16))) __bf16 v16bf;
typedef __attribute__((ext_vector_type(8)))  float  v8f;
typedef __attribute__((ext_vector_type(4)))  int    v4i_;

union Frag { v16bf v; uint4 u[2]; };

__device__ inline v8f wmma_bf16(const Frag& a, const Frag& b, v8f c) {
  // 8 args: (neg_a, A, neg_b, B, c_mod, C, reuse_a, reuse_b)
  return __builtin_amdgcn_wmma_f32_16x16x32_bf16(false, a.v, false, b.v,
                                                 (short)0, c, false, false);
}

// ---- async global->LDS staging (gfx1250 ASYNCcnt path), guarded fallback ----
#if __has_builtin(__builtin_amdgcn_global_load_async_to_lds_b128) && \
    __has_builtin(__builtin_amdgcn_s_wait_asynccnt)
#define HAS_ASYNC_LDS 1
typedef __attribute__((address_space(1))) v4i_ as1_v4i;
typedef __attribute__((address_space(3))) v4i_ as3_v4i;
#endif

__device__ inline void stage_b128(void* lds_dst, const void* g_src) {
#ifdef HAS_ASYNC_LDS
  __builtin_amdgcn_global_load_async_to_lds_b128((as1_v4i*)g_src, (as3_v4i*)lds_dst, 0, 0);
#else
  *(uint4*)lds_dst = *(const uint4*)g_src;
#endif
}
__device__ inline void stage_fence() {
#ifdef HAS_ASYNC_LDS
  __builtin_amdgcn_s_wait_asynccnt(0);
#endif
  __syncthreads();
}

__device__ inline unsigned short f2bf(float f) {
  union { float f; unsigned u; } x; x.f = f;
  unsigned r = x.u + 0x7FFFu + ((x.u >> 16) & 1u);   // round-to-nearest-even
  return (unsigned short)(r >> 16);
}

// ---------------------------------------------------------------------------
// Weight pack: fp32 [CO][CI][3][3] -> bf16 [CO][KP], K padded with zeros
// ---------------------------------------------------------------------------
__global__ void k_packw(const float* __restrict__ w, unsigned short* __restrict__ out,
                        int CO, int CI, int KP) {
  long total = (long)CO * KP;
  int K9 = CI * 9;
  for (long t = blockIdx.x * (long)blockDim.x + threadIdx.x; t < total;
       t += (long)gridDim.x * blockDim.x) {
    int kp = (int)(t % KP);
    long co = t / KP;
    unsigned short v = 0;
    if (kp < K9) v = f2bf(w[co * K9 + kp]);
    out[t] = v;
  }
}

// ---------------------------------------------------------------------------
// im2col (SAME, 3x3, stride 1) -> bf16 Col[(b*H*W + p)*KP + k]
// ---------------------------------------------------------------------------
__global__ void k_im2col(const float* __restrict__ X, long bStride, int C, int H, int W,
                         int KP, unsigned short* __restrict__ Col, int B) {
  long HW = (long)H * W;
  long total = (long)B * HW * KP;
  int K9 = C * 9;
  for (long t = blockIdx.x * (long)blockDim.x + threadIdx.x; t < total;
       t += (long)gridDim.x * blockDim.x) {
    int k = (int)(t % KP);
    long rest = t / KP;
    int p = (int)(rest % HW);
    int b = (int)(rest / HW);
    unsigned short v = 0;
    if (k < K9) {
      int ci = k / 9, r = k % 9, ky = r / 3, kx = r % 3;
      int y = p / W + ky - 1, x = p % W + kx - 1;
      if (y >= 0 && y < H && x >= 0 && x < W)
        v = f2bf(X[(long)b * bStride + (long)ci * HW + (long)y * W + x]);
    }
    Col[t] = v;
  }
}

// ---------------------------------------------------------------------------
// Implicit-GEMM conv: Out[b][oc][p] = relu(sum_k W[oc][k]*Col[b][p][k] + bias)
// 256 thr = 8 waves (2x4); block tile 64(OC) x 128(P); each wave owns 2x2
// 16x16 tiles -> 4 WMMA per K-stage of 32.
// ---------------------------------------------------------------------------
__global__ __launch_bounds__(256)
void k_gemm_bias_relu(const unsigned short* __restrict__ Wm,   // [OC][K] bf16
                      const unsigned short* __restrict__ Col,  // [B][P][K] bf16
                      const float* __restrict__ bias,
                      float* __restrict__ Out,
                      int OC, int P, int K,
                      long bStrideCol, long bStrideOut, long ocStrideOut) {
  __shared__ unsigned short As[64 * 32];   // 4 KB
  __shared__ unsigned short Bs[128 * 32];  // 8 KB
  int b = blockIdx.z;
  int ocBase = blockIdx.y * 64;
  int pBase  = blockIdx.x * 128;
  const unsigned short* colB = Col + (long)b * bStrideCol;
  float* outB = Out + (long)b * bStrideOut;
  int tid = threadIdx.x, wave = tid >> 5, lane = tid & 31;
  int wr = wave >> 2, wc = wave & 3;
  int mn = lane & 15, g = lane >> 4;
  v8f acc00 = {}, acc01 = {}, acc10 = {}, acc11 = {};
  for (int k0 = 0; k0 < K; k0 += 32) {
    __syncthreads();   // previous-stage fragment reads complete before overwrite
    for (int idx = tid; idx < 768; idx += 256) {
      if (idx < 256) {            // A: 64 rows x 32 k
        int row = idx >> 2, c = idx & 3;
        stage_b128(&As[row * 32 + c * 8], &Wm[(long)(ocBase + row) * K + k0 + c * 8]);
      } else {                    // B: 128 rows x 32 k
        int j = idx - 256;
        int row = j >> 2, c = j & 3;
        if (pBase + row < P)
          stage_b128(&Bs[row * 32 + c * 8], &colB[(long)(pBase + row) * K + k0 + c * 8]);
        else
          *(uint4*)&Bs[row * 32 + c * 8] = make_uint4(0u, 0u, 0u, 0u);
      }
    }
    stage_fence();
    Frag a0, a1, b0, b1;
    const unsigned short* ar0 = &As[(32 * wr + mn) * 32];
    const unsigned short* ar1 = &As[(32 * wr + 16 + mn) * 32];
    a0.u[0] = *(const uint4*)&ar0[16 * g];  a0.u[1] = *(const uint4*)&ar0[16 * g + 8];
    a1.u[0] = *(const uint4*)&ar1[16 * g];  a1.u[1] = *(const uint4*)&ar1[16 * g + 8];
    const unsigned short* br0 = &Bs[(32 * wc + mn) * 32];
    const unsigned short* br1 = &Bs[(32 * wc + 16 + mn) * 32];
    b0.u[0] = *(const uint4*)&br0[16 * g]; b0.u[1] = *(const uint4*)&br0[16 * g + 8];
    b1.u[0] = *(const uint4*)&br1[16 * g]; b1.u[1] = *(const uint4*)&br1[16 * g + 8];
    acc00 = wmma_bf16(a0, b0, acc00);
    acc01 = wmma_bf16(a0, b1, acc01);
    acc10 = wmma_bf16(a1, b0, acc10);
    acc11 = wmma_bf16(a1, b1, acc11);
  }
  // Epilogue: C layout — VGPR r holds row (r + 8g), col = lane&15
  auto store_tile = [&](v8f acc, int rOff, int cOff) {
#pragma unroll
    for (int r = 0; r < 8; ++r) {
      int oc = ocBase + 32 * wr + rOff + r + 8 * g;
      int p  = pBase + 32 * wc + cOff + mn;
      if (p < P) {
        float v = acc[r] + bias[oc];
        outB[(long)oc * ocStrideOut + p] = v > 0.f ? v : 0.f;
      }
    }
  };
  store_tile(acc00, 0, 0);   store_tile(acc01, 0, 16);
  store_tile(acc10, 16, 0);  store_tile(acc11, 16, 16);
}

// ---------------------------------------------------------------------------
// 2x2 max pool, stride 2
// ---------------------------------------------------------------------------
__global__ void k_pool(const float* __restrict__ in, float* __restrict__ out,
                       long BC, int H, int W) {
  int Ho = H >> 1, Wo = W >> 1;
  long total = BC * (long)Ho * Wo;
  for (long t = blockIdx.x * (long)blockDim.x + threadIdx.x; t < total;
       t += (long)gridDim.x * blockDim.x) {
    int xo = (int)(t % Wo);
    long r = t / Wo;
    int yo = (int)(r % Ho);
    long bc = r / Ho;
    const float* p = in + (bc * H + yo * 2) * (long)W + xo * 2;
    out[t] = fmaxf(fmaxf(p[0], p[1]), fmaxf(p[W], p[W + 1]));
  }
}

// ---------------------------------------------------------------------------
// Bicubic (align_corners=True, A=-0.75) upsample into feat channels [chanOff..)
// ---------------------------------------------------------------------------
__device__ inline float cubw(float t) {
  const float A = -0.75f;
  float at = fabsf(t);
  if (at <= 1.f) return ((A + 2.f) * at - (A + 3.f)) * at * at + 1.f;
  if (at < 2.f)  return ((A * at - 5.f * A) * at + 8.f * A) * at - 4.f * A;
  return 0.f;
}

__global__ void k_bicubic(const float* __restrict__ in, long inBStride, int C, int h, int w,
                          float* __restrict__ feat, long featBStride, int chanOff,
                          int H, int W, int B) {
  long total = (long)B * C * H * W;
  for (long t = blockIdx.x * (long)blockDim.x + threadIdx.x; t < total;
       t += (long)gridDim.x * blockDim.x) {
    int x = (int)(t % W);
    long r = t / W;
    int y = (int)(r % H); r /= H;
    int c = (int)(r % C);
    int b = (int)(r / C);
    float py = (float)y * (float)(h - 1) / (float)(H - 1);
    float px = (float)x * (float)(w - 1) / (float)(W - 1);
    int iy = (int)floorf(py), ix = (int)floorf(px);
    float fy = py - iy, fx = px - ix;
    float wy[4], wx[4];
#pragma unroll
    for (int o = 0; o < 4; ++o) { wy[o] = cubw(fy - (float)(o - 1)); wx[o] = cubw(fx - (float)(o - 1)); }
    const float* src = in + (long)b * inBStride + (long)c * h * w;
    float s = 0.f;
#pragma unroll
    for (int oy = 0; oy < 4; ++oy) {
      int yy = iy - 1 + oy; yy = yy < 0 ? 0 : (yy > h - 1 ? h - 1 : yy);
      float ra = 0.f;
#pragma unroll
      for (int ox = 0; ox < 4; ++ox) {
        int xx = ix - 1 + ox; xx = xx < 0 ? 0 : (xx > w - 1 ? w - 1 : xx);
        ra += wx[ox] * src[(long)yy * w + xx];
      }
      s += wy[oy] * ra;
    }
    feat[(long)b * featBStride + (long)(chanOff + c) * H * W + (long)y * W + x] = s;
  }
}

// ---------------------------------------------------------------------------
// Per (b,c) mean over concat spatial of both feature maps (std==mean bug kept)
// ---------------------------------------------------------------------------
__global__ void k_mean(const float* __restrict__ fr, const float* __restrict__ ft,
                       float* __restrict__ mean, long Pr, long Pt, int C) {
  int c = blockIdx.x, b = blockIdx.y;
  const float* a = fr + ((long)b * C + c) * Pr;
  const float* t2 = ft + ((long)b * C + c) * Pt;
  float s = 0.f;
  for (long i = threadIdx.x; i < Pr; i += blockDim.x) s += a[i];
  for (long i = threadIdx.x; i < Pt; i += blockDim.x) s += t2[i];
  __shared__ float red[256];
  red[threadIdx.x] = s;
  __syncthreads();
  for (int off = 128; off > 0; off >>= 1) {
    if ((int)threadIdx.x < off) red[threadIdx.x] += red[threadIdx.x + off];
    __syncthreads();
  }
  if (threadIdx.x == 0) mean[(long)b * C + c] = red[0] / (float)(Pr + Pt);
}

// ---------------------------------------------------------------------------
// Normalize + transpose to bf16: outT[b][i][c] = (feat[b][c][i]-m)/(m+1e-8)
// ---------------------------------------------------------------------------
__global__ void k_normT(const float* __restrict__ feat, const float* __restrict__ mean,
                        unsigned short* __restrict__ outT, long P, int C, int B) {
  long total = (long)B * P * C;
  for (long t = blockIdx.x * (long)blockDim.x + threadIdx.x; t < total;
       t += (long)gridDim.x * blockDim.x) {
    int c = (int)(t % C);
    long i = (t / C) % P;
    int b = (int)(t / (P * (long)C));
    float m = mean[(long)b * C + c];
    float v = (feat[((long)b * C + c) * P + i] - m) / (m + 1e-8f);
    outT[t] = f2bf(v);
  }
}

// ---------------------------------------------------------------------------
// Correlation row-stat pass (WMMA, K=320 fully unrolled, no barriers in K).
//  A: [B][PA][320] bf16 rows (LDS-resident strip of 16), Bm: [B][PB][320].
//  B fragments are loaded straight from global (L2-resident matrices).
//  Each wave owns 2 j-tiles -> strip of 256 columns per iteration.
//  mode 0: stat[b][i]  = max_j  A_i . B_j
//  mode 1: stat2[b][i] = sum_j exp(coef * (A_i.B_j - stat[b][i]))
// ---------------------------------------------------------------------------
__global__ __launch_bounds__(256)
void k_corr_stat(const unsigned short* __restrict__ A, const unsigned short* __restrict__ Bm,
                 const float* __restrict__ coefPtr, float* __restrict__ stat,
                 float* __restrict__ stat2, int PA, int PB, int mode) {
  __shared__ unsigned short As[16 * 320];  // 10 KB
  __shared__ float Cs[16 * 256];           // 16 KB
  int b = blockIdx.y;
  int iBase = blockIdx.x * 16;
  const unsigned short* Ab = A + (long)b * PA * 320;
  const unsigned short* Bb = Bm + (long)b * PB * 320;
  int tid = threadIdx.x, wave = tid >> 5, lane = tid & 31;
  int mn = lane & 15, g = lane >> 4;
  for (int idx = tid; idx < 640; idx += 256) {
    int row = idx / 40, c = idx % 40;
    stage_b128(&As[row * 320 + c * 8], &Ab[(long)(iBase + row) * 320 + c * 8]);
  }
  stage_fence();
  float coef = *coefPtr;
  float baseMax = (mode == 1 && tid < 16) ? stat[(long)b * PA + iBase + tid] : 0.f;
  float rmax = -3.4e38f, rsum = 0.f;
  for (int jBase = 0; jBase < PB; jBase += 256) {
    v8f acc0 = {}, acc1 = {};
    long j0 = jBase + (wave << 5) + mn;
    long j1 = j0 + 16;
#pragma unroll
    for (int k0 = 0; k0 < 320; k0 += 32) {
      Frag a, b0, b1;
      const unsigned short* ar = &As[mn * 320 + k0];
      a.u[0] = *(const uint4*)&ar[16 * g];
      a.u[1] = *(const uint4*)&ar[16 * g + 8];
      const unsigned short* br0 = &Bb[j0 * 320 + k0 + 16 * g];
      b0.u[0] = *(const uint4*)&br0[0];
      b0.u[1] = *(const uint4*)&br0[8];
      const unsigned short* br1 = &Bb[j1 * 320 + k0 + 16 * g];
      b1.u[0] = *(const uint4*)&br1[0];
      b1.u[1] = *(const uint4*)&br1[8];
      acc0 = wmma_bf16(a, b0, acc0);
      acc1 = wmma_bf16(a, b1, acc1);
    }
    __syncthreads();   // previous scan finished before Cs overwrite
#pragma unroll
    for (int r = 0; r < 8; ++r) {
      Cs[(r + 8 * g) * 256 + (wave << 5) + mn]      = acc0[r];
      Cs[(r + 8 * g) * 256 + (wave << 5) + 16 + mn] = acc1[r];
    }
    __syncthreads();
    if (tid < 16) {
      const float* row = &Cs[tid * 256];
      if (mode == 0) {
        for (int jj = 0; jj < 256; ++jj) rmax = fmaxf(rmax, row[jj]);
      } else {
        for (int jj = 0; jj < 256; ++jj) rsum += expf(coef * (row[jj] - baseMax));
      }
    }
  }
  if (tid < 16) {
    if (mode == 0) stat[(long)b * PA + iBase + tid] = rmax;
    else           stat2[(long)b * PA + iBase + tid] = rsum;
  }
}

// ---------------------------------------------------------------------------
// Final pass: confidence = sqrt(softmax_col * softmax_row), top-10 mean / row
// ---------------------------------------------------------------------------
__global__ __launch_bounds__(256)
void k_corr_topk(const unsigned short* __restrict__ A, const unsigned short* __restrict__ Bm,
                 const float* __restrict__ crPtr, const float* __restrict__ ctPtr,
                 const float* __restrict__ rmax, const float* __restrict__ rsum,
                 const float* __restrict__ cmax, const float* __restrict__ csum,
                 float* __restrict__ out, int PA, int PB) {
  __shared__ unsigned short As[16 * 320];
  __shared__ float Cs[16 * 256];
  int b = blockIdx.y;
  int iBase = blockIdx.x * 16;
  const unsigned short* Ab = A + (long)b * PA * 320;
  const unsigned short* Bb = Bm + (long)b * PB * 320;
  int tid = threadIdx.x, wave = tid >> 5, lane = tid & 31;
  int mn = lane & 15, g = lane >> 4;
  for (int idx = tid; idx < 640; idx += 256) {
    int row = idx / 40, c = idx % 40;
    stage_b128(&As[row * 320 + c * 8], &Ab[(long)(iBase + row) * 320 + c * 8]);
  }
  stage_fence();
  float cr = *crPtr, ct = *ctPtr;
  float mi = 0.f, lsi = 0.f;
  if (tid < 16) {
    mi  = rmax[(long)b * PA + iBase + tid];
    lsi = logf(rsum[(long)b * PA + iBase + tid]);
  }
  float tk[10];
#pragma unroll
  for (int p = 0; p < 10; ++p) tk[p] = -3.4e38f;
  for (int jBase = 0; jBase < PB; jBase += 256) {
    v8f acc0 = {}, acc1 = {};
    long j0 = jBase + (wave << 5) + mn;
    long j1 = j0 + 16;
#pragma unroll
    for (int k0 = 0; k0 < 320; k0 += 32) {
      Frag a, b0, b1;
      const unsigned short* ar = &As[mn * 320 + k0];
      a.u[0] = *(const uint4*)&ar[16 * g];
      a.u[1] = *(const uint4*)&ar[16 * g + 8];
      const unsigned short* br0 = &Bb[j0 * 320 + k0 + 16 * g];
      b0.u[0] = *(const uint4*)&br0[0];
      b0.u[1] = *(const uint4*)&br0[8];
      const unsigned short* br1 = &Bb[j1 * 320 + k0 + 16 * g];
      b1.u[0] = *(const uint4*)&br1[0];
      b1.u[1] = *(const uint4*)&br1[8];
      acc0 = wmma_bf16(a, b0, acc0);
      acc1 = wmma_bf16(a, b1, acc1);
    }
    __syncthreads();
#pragma unroll
    for (int r = 0; r < 8; ++r) {
      Cs[(r + 8 * g) * 256 + (wave << 5) + mn]      = acc0[r];
      Cs[(r + 8 * g) * 256 + (wave << 5) + 16 + mn] = acc1[r];
    }
    __syncthreads();
    if (tid < 16) {
      const float* row = &Cs[tid * 256];
      for (int jj = 0; jj < 256; ++jj) {
        int j = jBase + jj;
        float d = row[jj];
        float cmj = cmax[(long)b * PB + j];
        float lsj = logf(csum[(long)b * PB + j]);
        float v = expf(0.5f * (cr * (d - cmj) + ct * (d - mi)) - 0.5f * (lsj + lsi));
        if (v > tk[9]) {
          tk[9] = v;
#pragma unroll
          for (int p = 9; p > 0; --p)
            if (tk[p] > tk[p - 1]) { float tt = tk[p - 1]; tk[p - 1] = tk[p]; tk[p] = tt; }
        }
      }
    }
  }
  if (tid < 16) {
    float s = 0.f;
#pragma unroll
    for (int p = 0; p < 10; ++p) s += tk[p];
    out[(long)b * PA + iBase + tid] = s * 0.1f;
  }
}

// ---------------------------------------------------------------------------
// Host orchestration
// ---------------------------------------------------------------------------
static inline int gsBlocks(long total) {
  long b = (total + 255) / 256;
  if (b > 65535) b = 65535;
  return (int)(b > 0 ? b : 1);
}

static void run_extract(const float* x, int B, int H, int W,
                        unsigned short* const* wpk, const float* const* bias,
                        const int* KPv,
                        unsigned short* col, float* actA, float* actB, float* feat,
                        hipStream_t stream) {
  int P = H * W, H2 = H / 2, W2 = W / 2, P2 = H2 * W2, H4 = H / 4, W4 = W / 4, P4 = H4 * W4;

  auto im2col = [&](const float* X, long bStride, int C, int h, int w, int kp) {
    long total = (long)B * h * w * kp;
    k_im2col<<<gsBlocks(total), 256, 0, stream>>>(X, bStride, C, h, w, kp, col, B);
  };
  auto gemm = [&](const unsigned short* Wm, const float* b_, float* outp,
                  int OC, int Pp, int K, long bStrideOut, long ocStride) {
    dim3 grd((Pp + 127) / 128, OC / 64, B);
    k_gemm_bias_relu<<<grd, 256, 0, stream>>>(Wm, col, b_, outp, OC, Pp, K,
                                              (long)Pp * K, bStrideOut, ocStride);
  };
  auto pool = [&](const float* in, float* outp, long BC, int h, int w) {
    k_pool<<<gsBlocks(BC * (h / 2) * (w / 2)), 256, 0, stream>>>(in, outp, BC, h, w);
  };

  // L0: x1 -> feat channels 0..63
  im2col(x, (long)3 * P, 3, H, W, KPv[0]);
  gemm(wpk[0], bias[0], feat, 64, P, KPv[0], (long)320 * P, P);
  // L1 + pool
  im2col(feat, (long)320 * P, 64, H, W, KPv[1]);
  gemm(wpk[1], bias[1], actA, 64, P, KPv[1], (long)64 * P, P);
  pool(actA, actB, (long)B * 64, H, W);
  // L2
  im2col(actB, (long)64 * P2, 64, H2, W2, KPv[2]);
  gemm(wpk[2], bias[2], actA, 128, P2, KPv[2], (long)128 * P2, P2);
  // L3 + pool
  im2col(actA, (long)128 * P2, 128, H2, W2, KPv[3]);
  gemm(wpk[3], bias[3], actB, 128, P2, KPv[3], (long)128 * P2, P2);
  pool(actB, actA, (long)B * 128, H2, W2);
  // L4..L7
  im2col(actA, (long)128 * P4, 128, H4, W4, KPv[4]);
  gemm(wpk[4], bias[4], actB, 256, P4, KPv[4], (long)256 * P4, P4);
  im2col(actB, (long)256 * P4, 256, H4, W4, KPv[5]);
  gemm(wpk[5], bias[5], actA, 256, P4, KPv[5], (long)256 * P4, P4);
  im2col(actA, (long)256 * P4, 256, H4, W4, KPv[6]);
  gemm(wpk[6], bias[6], actB, 256, P4, KPv[6], (long)256 * P4, P4);
  im2col(actB, (long)256 * P4, 256, H4, W4, KPv[7]);
  gemm(wpk[7], bias[7], actA, 256, P4, KPv[7], (long)256 * P4, P4);
  // bicubic upsample -> feat channels 64..319
  k_bicubic<<<gsBlocks((long)B * 256 * H * W), 256, 0, stream>>>(
      actA, (long)256 * P4, 256, H4, W4, feat, (long)320 * P, 64, H, W, B);
}

extern "C" void kernel_launch(void* const* d_in, const int* in_sizes, int n_in,
                              void* d_out, int out_size, void* d_ws, size_t ws_size,
                              hipStream_t stream) {
  (void)in_sizes; (void)n_in; (void)out_size;
  const float* w[8]; const float* bias[8];
  for (int i = 0; i < 8; ++i) { w[i] = (const float*)d_in[2 * i]; bias[i] = (const float*)d_in[2 * i + 1]; }
  const float* x_ref    = (const float*)d_in[16];
  const float* x_tmp    = (const float*)d_in[17];
  const float* coef_ref = (const float*)d_in[18];
  const float* coef_tmp = (const float*)d_in[19];
  float* out = (float*)d_out;

  const int COv[8] = {64, 64, 128, 128, 256, 256, 256, 256};
  const int CIv[8] = {3, 64, 64, 128, 128, 256, 256, 256};
  const int KPv[8] = {32, 576, 576, 1152, 1152, 2304, 2304, 2304};

  // ---- workspace bump allocation (256B aligned) ----
  char* ws = (char*)d_ws; size_t off = 0;
  auto alloc = [&](size_t bytes) -> void* {
    void* p = ws + off;
    off = (off + bytes + 255) & ~(size_t)255;
    return p;
  };
  unsigned short* wpk[8];
  for (int i = 0; i < 8; ++i) wpk[i] = (unsigned short*)alloc((size_t)COv[i] * KPv[i] * 2);
  unsigned short* col  = (unsigned short*)alloc(2ull * 9216 * 576 * 2);  // largest im2col
  float* actA  = (float*)alloc(2ull * 64 * 9216 * 4);
  float* actB  = (float*)alloc(2ull * 64 * 9216 * 4);
  float* featR = (float*)alloc(2ull * 320 * 9216 * 4);
  float* featT = (float*)alloc(2ull * 320 * 2304 * 4);
  unsigned short* FrT = (unsigned short*)alloc(2ull * 9216 * 320 * 2);
  unsigned short* FtT = (unsigned short*)alloc(2ull * 2304 * 320 * 2);
  float* meanA = (float*)alloc(2ull * 320 * 4);
  float* rmax = (float*)alloc(2ull * 9216 * 4);
  float* rsum = (float*)alloc(2ull * 9216 * 4);
  float* cmax = (float*)alloc(2ull * 2304 * 4);
  float* csum = (float*)alloc(2ull * 2304 * 4);
  if (off > ws_size) return;  // workspace too small; nothing safe to do

  // ---- pack weights to bf16 [CO][KP] ----
  for (int i = 0; i < 8; ++i) {
    long total = (long)COv[i] * KPv[i];
    k_packw<<<gsBlocks(total), 256, 0, stream>>>(w[i], wpk[i], COv[i], CIv[i], KPv[i]);
  }

  // ---- VGG feature extraction (both inputs) ----
  run_extract(x_ref, 2, 96, 96, wpk, bias, KPv, col, actA, actB, featR, stream);
  run_extract(x_tmp, 2, 48, 48, wpk, bias, KPv, col, actA, actB, featT, stream);

  // ---- normalizer (mean over concat spatial; std==mean per source bug) ----
  k_mean<<<dim3(320, 2), 256, 0, stream>>>(featR, featT, meanA, 9216, 2304, 320);
  k_normT<<<gsBlocks(2ll * 9216 * 320), 256, 0, stream>>>(featR, meanA, FrT, 9216, 320, 2);
  k_normT<<<gsBlocks(2ll * 2304 * 320), 256, 0, stream>>>(featT, meanA, FtT, 2304, 320, 2);

  // ---- correlation softmax stats (dist recomputed via WMMA, never stored) ----
  k_corr_stat<<<dim3(9216 / 16, 2), 256, 0, stream>>>(FrT, FtT, coef_tmp, rmax, rsum, 9216, 2304, 0);
  k_corr_stat<<<dim3(2304 / 16, 2), 256, 0, stream>>>(FtT, FrT, coef_ref, cmax, csum, 2304, 9216, 0);
  k_corr_stat<<<dim3(9216 / 16, 2), 256, 0, stream>>>(FrT, FtT, coef_tmp, rmax, rsum, 9216, 2304, 1);
  k_corr_stat<<<dim3(2304 / 16, 2), 256, 0, stream>>>(FtT, FrT, coef_ref, cmax, csum, 2304, 9216, 1);

  // ---- confidence + top-10 mean -> out [2,96,96,1] ----
  k_corr_topk<<<dim3(9216 / 16, 2), 256, 0, stream>>>(FrT, FtT, coef_ref, coef_tmp,
                                                      rmax, rsum, cmax, csum, out, 9216, 2304);
}
